// CausalSelfAttention_7919919694237
// MI455X (gfx1250) — matrix operand order repaired and checked by reading, not compiled, not run on hardware
//
#include <hip/hip_runtime.h>
#include <hip/hip_bf16.h>
#include <math.h>

typedef __bf16 bf16_t;
typedef __attribute__((ext_vector_type(16))) __bf16 v16bf;
typedef __attribute__((ext_vector_type(8)))  __bf16 v8bf;
typedef __attribute__((ext_vector_type(8)))  float  v8f;

#define WMMA_BF16(A, B, C) \
  __builtin_amdgcn_wmma_f32_16x16x32_bf16(false, (A), false, (B), (short)0, (C), false, false)

constexpr int D_MODEL  = 1024;
constexpr int N_HEADS  = 16;
constexpr int HEAD_DIM = 64;
constexpr int BATCH    = 4;
constexpr int SEQ      = 2048;
constexpr int ROWS     = BATCH * SEQ;   // 8192

// ---------------------------------------------------------------------------
// gfx1250 async global->LDS path (guarded so either toolchain compiles)
// builtin signature: (v4i addrspace(1)*, v4i addrspace(3)*, imm offset, imm cpol)
// ---------------------------------------------------------------------------
#if defined(__gfx1250__) && __has_builtin(__builtin_amdgcn_global_load_async_to_lds_b128)
#define HAVE_ASYNC_LDS 1
#else
#define HAVE_ASYNC_LDS 0
#endif

typedef int v4i_gcc __attribute__((vector_size(16)));
typedef __attribute__((address_space(1))) v4i_gcc g_v4i;
typedef __attribute__((address_space(3))) v4i_gcc l_v4i;

// copy 8 bf16 (16 bytes) global -> LDS
__device__ __forceinline__ void copy16_g2l(bf16_t* dst_lds, const bf16_t* src_g) {
#if HAVE_ASYNC_LDS
  __builtin_amdgcn_global_load_async_to_lds_b128((g_v4i*)src_g, (l_v4i*)dst_lds, 0, 0);
#else
  *(v8bf*)dst_lds = *(const v8bf*)src_g;
#endif
}

__device__ __forceinline__ void wait_async_lds() {
#if HAVE_ASYNC_LDS
#if __has_builtin(__builtin_amdgcn_s_wait_asynccnt)
  __builtin_amdgcn_s_wait_asynccnt(0);
#else
  asm volatile("s_wait_asynccnt 0" ::: "memory");
#endif
#endif
}

// ---------------------------------------------------------------------------
// helpers
// ---------------------------------------------------------------------------
__device__ __forceinline__ v16bf concat8(v8bf lo, v8bf hi) {
  return __builtin_shufflevector(lo, hi, 0,1,2,3,4,5,6,7,8,9,10,11,12,13,14,15);
}

// A-fragment (16x32 bf16, MxK), contiguous k order:
// lane 0-15 : M=lane,    elems = K[0..15]
// lane 16-31: M=lane-16, elems = K[16..31]
// p must already point at (row*ld + (hi?16:0)); 16B aligned.
__device__ __forceinline__ v16bf load_afrag(const bf16_t* p) {
  v8bf lo = *(const v8bf*)p;
  v8bf hi = *(const v8bf*)(p + 8);
  return concat8(lo, hi);
}

// ---------------------------------------------------------------------------
// fp32 -> bf16 conversion (grid-stride)
// ---------------------------------------------------------------------------
__global__ __launch_bounds__(256)
void cvt_f32_bf16(const float* __restrict__ src, bf16_t* __restrict__ dst, int n) {
  int i = blockIdx.x * blockDim.x + threadIdx.x;
  int stride = gridDim.x * blockDim.x;
  for (; i < n; i += stride) dst[i] = (bf16_t)src[i];
}

// ---------------------------------------------------------------------------
// bf16 GEMM: C[M,N] = A[M,K] * B[K,N] + bias
// MODE 0: store fp32 row-major into Cout
// MODE 1: QKV epilogue -> scatter bf16 into Q/K/V [B,H,T,Dh], Q scaled 0.125
// block: 256 threads = 8 waves; tile 128x128; wave tile 32x64; K step 64
// ---------------------------------------------------------------------------
template <int MODE>
__global__ __launch_bounds__(256)
void gemm_bf16(const bf16_t* __restrict__ A, const bf16_t* __restrict__ B,
               const float* __restrict__ bias, int M, int N, int K,
               float* __restrict__ Cout,
               bf16_t* __restrict__ Qo, bf16_t* __restrict__ Ko,
               bf16_t* __restrict__ Vo) {
  constexpr int LDA = 72;                       // padded (64+8) elem stride
  __shared__ __align__(32) bf16_t As[128 * LDA];  // [m][k] row-major
  __shared__ __align__(32) bf16_t Bt[128 * 64];   // transposed: [n][k]

  const int lane = threadIdx.x & 31;
  const int wid  = threadIdx.x >> 5;
  const int wr   = wid >> 1;                    // 0..3  (M dir)
  const int wc   = wid & 1;                     // 0..1  (N dir)
  const int hi   = (lane >= 16) ? 1 : 0;
  const int coll = lane & 15;
  const int m0 = blockIdx.y * 128;
  const int n0 = blockIdx.x * 128;

  const v8f vzero = {0.f,0.f,0.f,0.f,0.f,0.f,0.f,0.f};
  v8f acc[2][4];
#pragma unroll
  for (int mt = 0; mt < 2; ++mt)
#pragma unroll
    for (int nt = 0; nt < 4; ++nt) acc[mt][nt] = vzero;

  const int arow  = threadIdx.x >> 1;           // 0..127
  const int ahalf = threadIdx.x & 1;            // 0..1   (32-elem half)
  const int bkk   = threadIdx.x & 63;           // 0..63  (k row of B)
  const int bng   = threadIdx.x >> 6;           // 0..3   (32-col group)

  for (int k0 = 0; k0 < K; k0 += 64) {
    __syncthreads();
    {   // stage A tile 128x64 via async b128 (4 x 16B per thread)
      const bf16_t* g = A + (size_t)(m0 + arow) * K + k0 + ahalf * 32;
      bf16_t* s = As + arow * LDA + ahalf * 32;
#pragma unroll
      for (int i = 0; i < 4; ++i) copy16_g2l(s + 8 * i, g + 8 * i);
    }
    {   // stage B tile 64x128, transposed into Bt[n][k]
      const bf16_t* g = B + (size_t)(k0 + bkk) * N + n0 + bng * 32;
      v16bf b0 = *(const v16bf*)(g);
      v16bf b1 = *(const v16bf*)(g + 16);
#pragma unroll
      for (int i = 0; i < 16; ++i) {
        Bt[(bng * 32 + i) * 64 + bkk]      = b0[i];
        Bt[(bng * 32 + 16 + i) * 64 + bkk] = b1[i];
      }
    }
    if (k0 + 64 < K) {   // prefetch next A/B tiles into L2
      __builtin_prefetch(A + (size_t)(m0 + arow) * K + k0 + 64 + ahalf * 32, 0, 1);
      __builtin_prefetch(B + (size_t)(k0 + 64 + bkk) * N + n0 + bng * 32, 0, 1);
    }
    wait_async_lds();
    __syncthreads();

    const int ak = hi ? 16 : 0;
    const int bk = hi ? 16 : 0;
    v16bf af[2][2];
#pragma unroll
    for (int mt = 0; mt < 2; ++mt)
#pragma unroll
      for (int c = 0; c < 2; ++c)
        af[mt][c] = load_afrag(As + (wr * 32 + mt * 16 + coll) * LDA + c * 32 + ak);
#pragma unroll
    for (int nt = 0; nt < 4; ++nt) {
      const bf16_t* bp = Bt + (wc * 64 + nt * 16 + coll) * 64 + bk;
      v16bf bf0 = *(const v16bf*)(bp);
      v16bf bf1 = *(const v16bf*)(bp + 32);
#pragma unroll
      for (int mt = 0; mt < 2; ++mt) {
        acc[mt][nt] = WMMA_BF16(af[mt][0], bf0, acc[mt][nt]);
        acc[mt][nt] = WMMA_BF16(af[mt][1], bf1, acc[mt][nt]);
      }
    }
  }

  // epilogue: C element r <-> row (r + 8*hi), col (lane&15)
#pragma unroll
  for (int mt = 0; mt < 2; ++mt) {
#pragma unroll
    for (int nt = 0; nt < 4; ++nt) {
#pragma unroll
      for (int r = 0; r < 8; ++r) {
        int gm = m0 + wr * 32 + mt * 16 + r + 8 * hi;
        int gn = n0 + wc * 64 + nt * 16 + coll;
        float v = acc[mt][nt][r] + bias[gn];
        if (MODE == 0) {
          Cout[(size_t)gm * N + gn] = v;
        } else {
          int which = gn >> 10;          // 0=q 1=k 2=v
          int cc = gn & 1023;
          int h = cc >> 6, d = cc & 63;
          int b = gm >> 11, t = gm & 2047;
          size_t idx = ((((size_t)b * N_HEADS + h) * SEQ) + t) * HEAD_DIM + d;
          if (which == 0) {
            Qo[idx] = (bf16_t)(v * 0.125f);   // fold 1/sqrt(Dh)
          } else if (which == 1) {
            Ko[idx] = (bf16_t)v;
          } else {
            Vo[idx] = (bf16_t)v;
          }
        }
      }
    }
  }
}

// ---------------------------------------------------------------------------
// flash attention (causal): Q,K,V bf16 [B,H,T,Dh] -> Y bf16 [B,T,H,Dh]
// block = 128 query rows (8 waves x 16 rows), key chunk = 32
// ---------------------------------------------------------------------------
__global__ __launch_bounds__(256)
void attn_fwd(const bf16_t* __restrict__ Q, const bf16_t* __restrict__ K,
              const bf16_t* __restrict__ V, bf16_t* __restrict__ Y) {
  __shared__ __align__(32) bf16_t Ks[32 * 64];      // [key][d]
  __shared__ __align__(32) bf16_t Vt[64 * 32];      // [d][key]
  __shared__ __align__(32) bf16_t Ps[8][16 * 32];   // per-wave P tile

  const int lane = threadIdx.x & 31;
  const int wid  = threadIdx.x >> 5;
  const int hi   = (lane >= 16) ? 1 : 0;
  const int coll = lane & 15;

  const int bh = blockIdx.x;                 // b*16 + h
  const int q0 = blockIdx.y * 128;
  const int b  = bh >> 4;
  const int h  = bh & 15;

  const bf16_t* qh = Q + (size_t)bh * SEQ * HEAD_DIM;
  const bf16_t* kh = K + (size_t)bh * SEQ * HEAD_DIM;
  const bf16_t* vh = V + (size_t)bh * SEQ * HEAD_DIM;
  const int qrow0 = q0 + wid * 16;

  // Q fragments (16 x 64 -> two 16x32 A-frags), straight from global
  v16bf qa[2];
  {
    int t  = qrow0 + coll;
    int ak = hi ? 16 : 0;
#pragma unroll
    for (int c = 0; c < 2; ++c)
      qa[c] = load_afrag(qh + (size_t)t * HEAD_DIM + c * 32 + ak);
  }

  const v8f vzero = {0.f,0.f,0.f,0.f,0.f,0.f,0.f,0.f};
  v8f o[4];
#pragma unroll
  for (int nt = 0; nt < 4; ++nt) o[nt] = vzero;
  float m[8], l[8];
#pragma unroll
  for (int r = 0; r < 8; ++r) { m[r] = -__builtin_inff(); l[r] = 0.f; }

  const int srow = threadIdx.x >> 3;   // 0..31  (kv row staged by this thread)
  const int sseg = threadIdx.x & 7;    // 0..7   (8-elem segment)

  const int nchunk = (q0 + 128) / 32;
  for (int j = 0; j < nchunk; ++j) {
    const int kv0 = j * 32;
    __syncthreads();
    {   // stage K (row-major, async b128) and V (register-transposed)
      copy16_g2l(Ks + srow * 64 + sseg * 8,
                 kh + (size_t)(kv0 + srow) * HEAD_DIM + sseg * 8);
      v8bf vv = *(const v8bf*)(vh + (size_t)(kv0 + srow) * HEAD_DIM + sseg * 8);
#pragma unroll
      for (int i = 0; i < 8; ++i) Vt[(sseg * 8 + i) * 32 + srow] = vv[i];
      if (j + 1 < nchunk) {
        __builtin_prefetch(kh + (size_t)(kv0 + 32 + srow) * HEAD_DIM + sseg * 8, 0, 1);
        __builtin_prefetch(vh + (size_t)(kv0 + 32 + srow) * HEAD_DIM + sseg * 8, 0, 1);
      }
    }
    wait_async_lds();
    __syncthreads();

    if (kv0 <= qrow0 + 15) {      // chunk intersects this wave's causal region
      // ---- S = Q * K^T : 16 x 32 in two accumulators
      v8f s0 = vzero, s1 = vzero;
      const int bk = hi ? 16 : 0;
#pragma unroll
      for (int c = 0; c < 2; ++c) {
        v16bf kb0 = *(const v16bf*)(Ks + (coll)      * 64 + c * 32 + bk);
        v16bf kb1 = *(const v16bf*)(Ks + (16 + coll) * 64 + c * 32 + bk);
        s0 = WMMA_BF16(qa[c], kb0, s0);
        s1 = WMMA_BF16(qa[c], kb1, s1);
      }

      // ---- causal mask + online softmax (row = r + 8*hi within wave tile)
      float osc[8];
#pragma unroll
      for (int r = 0; r < 8; ++r) {
        int qr = qrow0 + r + 8 * hi;
        float x0 = ((kv0 + coll)      > qr) ? -__builtin_inff() : s0[r];
        float x1 = ((kv0 + 16 + coll) > qr) ? -__builtin_inff() : s1[r];
        float mx = fmaxf(x0, x1);
#pragma unroll
        for (int d = 1; d < 16; d <<= 1) mx = fmaxf(mx, __shfl_xor(mx, d, 32));
        float mnew = fmaxf(m[r], mx);
        float sc = __expf(m[r] - mnew);
        float p0 = __expf(x0 - mnew);
        float p1 = __expf(x1 - mnew);
        float ps = p0 + p1;
#pragma unroll
        for (int d = 1; d < 16; d <<= 1) ps += __shfl_xor(ps, d, 32);
        l[r] = l[r] * sc + ps;
        m[r] = mnew;
        osc[r] = sc;
        // C-layout -> row-major P in per-wave LDS (bf16)
        Ps[wid][(r + 8 * hi) * 32 + coll]      = (bf16_t)p0;
        Ps[wid][(r + 8 * hi) * 32 + coll + 16] = (bf16_t)p1;
      }
#pragma unroll
      for (int nt = 0; nt < 4; ++nt)
#pragma unroll
        for (int r = 0; r < 8; ++r) o[nt][r] *= osc[r];

      // ---- O += P * V : reload P as A-frag, V^T as B-frags
      v16bf pa;
      {
        int ak = hi ? 16 : 0;
        pa = load_afrag(&Ps[wid][coll * 32 + ak]);
      }
#pragma unroll
      for (int nt = 0; nt < 4; ++nt) {
        v16bf vb = *(const v16bf*)(Vt + (nt * 16 + coll) * 32 + bk);
        o[nt] = WMMA_BF16(pa, vb, o[nt]);
      }
    }
  }

  // ---- normalize and store Y [B,T,H,Dh]
#pragma unroll
  for (int r = 0; r < 8; ++r) {
    float inv = (l[r] > 0.f) ? (1.f / l[r]) : 0.f;
    int t = qrow0 + r + 8 * hi;
#pragma unroll
    for (int nt = 0; nt < 4; ++nt) {
      int d = nt * 16 + coll;
      size_t idx = (((size_t)b * SEQ + t) * N_HEADS + h) * HEAD_DIM + d;
      Y[idx] = (bf16_t)(o[nt][r] * inv);
    }
  }
}

// ---------------------------------------------------------------------------
// launcher
// ---------------------------------------------------------------------------
extern "C" void kernel_launch(void* const* d_in, const int* in_sizes, int n_in,
                              void* d_out, int out_size, void* d_ws, size_t ws_size,
                              hipStream_t stream) {
  const float* x     = (const float*)d_in[0];
  const float* w_qkv = (const float*)d_in[1];
  const float* b_qkv = (const float*)d_in[2];
  const float* w_out = (const float*)d_in[3];
  const float* b_out = (const float*)d_in[4];
  float* out = (float*)d_out;

  // workspace layout (bf16)
  char* p = (char*)d_ws;
  bf16_t* xb    = (bf16_t*)p; p += (size_t)ROWS * D_MODEL * sizeof(bf16_t);
  bf16_t* wqkvb = (bf16_t*)p; p += (size_t)D_MODEL * 3 * D_MODEL * sizeof(bf16_t);
  bf16_t* woutb = (bf16_t*)p; p += (size_t)D_MODEL * D_MODEL * sizeof(bf16_t);
  bf16_t* q     = (bf16_t*)p; p += (size_t)ROWS * D_MODEL * sizeof(bf16_t);
  bf16_t* k     = (bf16_t*)p; p += (size_t)ROWS * D_MODEL * sizeof(bf16_t);
  bf16_t* v     = (bf16_t*)p; p += (size_t)ROWS * D_MODEL * sizeof(bf16_t);
  bf16_t* y     = (bf16_t*)p; p += (size_t)ROWS * D_MODEL * sizeof(bf16_t);

  cvt_f32_bf16<<<4096, 256, 0, stream>>>(x, xb, ROWS * D_MODEL);
  cvt_f32_bf16<<<4096, 256, 0, stream>>>(w_qkv, wqkvb, D_MODEL * 3 * D_MODEL);
  cvt_f32_bf16<<<2048, 256, 0, stream>>>(w_out, woutb, D_MODEL * D_MODEL);

  // QKV projection: [8192,1024] x [1024,3072] -> scattered q/k/v
  dim3 g1(3 * D_MODEL / 128, ROWS / 128);
  gemm_bf16<1><<<g1, 256, 0, stream>>>(xb, wqkvb, b_qkv, ROWS, 3 * D_MODEL, D_MODEL,
                                       nullptr, q, k, v);

  // attention: 64 (b,h) pairs x 16 query chunks of 128
  dim3 g2(BATCH * N_HEADS, SEQ / 128);
  attn_fwd<<<g2, 256, 0, stream>>>(q, k, v, y);

  // output projection: [8192,1024] x [1024,1024] -> fp32 out
  dim3 g3(D_MODEL / 128, ROWS / 128);
  gemm_bf16<0><<<g3, 256, 0, stream>>>(y, woutb, b_out, ROWS, D_MODEL, D_MODEL,
                                       out, nullptr, nullptr, nullptr);
}
